// CGCNN_Interactions_85993835200799
// MI455X (gfx1250) — matrix operand order repaired and checked by reading, not compiled, not run on hardware
//
#include <hip/hip_runtime.h>

#define N_NODES 10000
#define N_EDGES 160000
#define HID 128
#define NG 100
#define NF 64
#define ET 64  // edges per block (4 WMMA M-tiles)

typedef __attribute__((ext_vector_type(16))) __bf16 v16bf;
typedef __attribute__((ext_vector_type(8)))  __bf16 v8bf;
typedef __attribute__((ext_vector_type(8)))  float  v8f;
typedef unsigned uint32x4 __attribute__((ext_vector_type(4)));
typedef unsigned uint32x8 __attribute__((ext_vector_type(8)));

__device__ __forceinline__ v8f wmma_bf16(v16bf a, v16bf b, v8f c) {
  // D = A(16x32) * B(32x16) + C, f32 accumulate
  return __builtin_amdgcn_wmma_f32_16x16x32_bf16(false, a, false, b, (short)0, c,
                                                 false, false);
}

// 16x32 A-fragment for this lane from an LDS row: two contiguous 16B runs
// (K = base..base+7 and base+16..base+23) -> 2x ds_load_b128.
__device__ __forceinline__ v16bf load_a_lds(const __bf16* row, int base) {
  v8bf lo = *(const v8bf*)(row + base);
  v8bf hi = *(const v8bf*)(row + base + 16);
  return __builtin_shufflevector(lo, hi, 0, 1, 2, 3, 4, 5, 6, 7, 8, 9, 10, 11,
                                 12, 13, 14, 15);
}

// Packed-B fragment: pack[((nt*KS+ks)*32+lane)*16 + j] -> one 32B load.
__device__ __forceinline__ v16bf load_b_pack(const __bf16* pack, int frag, int l) {
  return *(const v16bf*)(pack + (((size_t)frag * 32 + l) << 4));
}

// ---------------- weight conversion + fragment packing (once per launch) -----
__global__ void convert_weights_kernel(const float* __restrict__ lin0_w,
                                       const float* __restrict__ nn_w1,
                                       const float* __restrict__ nn_w2,
                                       const float* __restrict__ root_w,
                                       __bf16* __restrict__ w0p,
                                       __bf16* __restrict__ w1p,
                                       __bf16* __restrict__ w2p,
                                       __bf16* __restrict__ rwp) {
  int i = blockIdx.x * blockDim.x + threadIdx.x;
  int stride = gridDim.x * blockDim.x;
  // lin0_w: [128][64] -> KS=4, NT=4
  for (int t = i; t < HID * NF; t += stride) {
    int j = t & 15, l = (t >> 4) & 31, f = t >> 9;
    int ks = f & 3, nt = f >> 2;
    int k = ks * 32 + ((l >= 16) ? 8 : 0) + ((j < 8) ? j : (8 + j));
    int n = nt * 16 + (l & 15);
    w0p[t] = (__bf16)lin0_w[k * NF + n];
  }
  // nn_w1: [100->128 pad][128] -> KS=4, NT=8
  for (int t = i; t < HID * HID; t += stride) {
    int j = t & 15, l = (t >> 4) & 31, f = t >> 9;
    int ks = f & 3, nt = f >> 2;
    int k = ks * 32 + ((l >= 16) ? 8 : 0) + ((j < 8) ? j : (8 + j));
    int n = nt * 16 + (l & 15);
    w1p[t] = (k < NG) ? (__bf16)nn_w1[k * HID + n] : (__bf16)0.f;
  }
  // nn_w2: [128][4096] -> KS=4, NT=256
  for (int t = i; t < HID * NF * NF; t += stride) {
    int j = t & 15, l = (t >> 4) & 31, f = t >> 9;
    int ks = f & 3, nt = f >> 2;
    int k = ks * 32 + ((l >= 16) ? 8 : 0) + ((j < 8) ? j : (8 + j));
    int n = nt * 16 + (l & 15);
    w2p[t] = (__bf16)nn_w2[k * (NF * NF) + n];
  }
  // root_w: [64][64] -> KS=2, NT=4
  for (int t = i; t < NF * NF; t += stride) {
    int j = t & 15, l = (t >> 4) & 31, f = t >> 9;
    int ks = f & 1, nt = f >> 1;
    int k = ks * 32 + ((l >= 16) ? 8 : 0) + ((j < 8) ? j : (8 + j));
    int n = nt * 16 + (l & 15);
    rwp[t] = (__bf16)root_w[k * NF + n];
  }
}

// edge_attr: [E][100] f32 -> [E][128] bf16, zero-padded (done once, used twice)
__global__ void convert_ea_kernel(const float* __restrict__ ea,
                                  __bf16* __restrict__ eaBF) {
  int i = blockIdx.x * blockDim.x + threadIdx.x;
  int stride = gridDim.x * blockDim.x;
  for (int t = i; t < N_EDGES * HID; t += stride) {
    int row = t >> 7, col = t & 127;
    float v = (col < NG) ? ea[(size_t)row * NG + col] : 0.f;
    eaBF[t] = (__bf16)v;
  }
}

__global__ void zero_kernel(float* __restrict__ p, int n) {
  int i = blockIdx.x * blockDim.x + threadIdx.x;
  if (i < n) p[i] = 0.f;
}

__global__ void degree_kernel(const long long* __restrict__ eidx,
                              float* __restrict__ deg) {
  int e = blockIdx.x * blockDim.x + threadIdx.x;
  if (e < N_EDGES) atomicAdd(&deg[(int)eidx[N_EDGES + e]], 1.0f);
}

__global__ void invdeg_kernel(float* __restrict__ deg) {
  int i = blockIdx.x * blockDim.x + threadIdx.x;
  if (i < N_NODES) deg[i] = 1.0f / fmaxf(deg[i], 1.0f);
}

// ---------------- lin0: x = relu(h @ lin0_w + b) ----------------
__global__ __launch_bounds__(128) void lin0_kernel(const float* __restrict__ h,
                                                   const __bf16* __restrict__ w0p,
                                                   const float* __restrict__ b0,
                                                   float* __restrict__ xout) {
  const int m0 = blockIdx.x * 16;
  const int wv = threadIdx.x >> 5;
  const int l = threadIdx.x & 31;
  const int lm = l & 15;
  const int hi = (l >= 16);
  const int n = wv * 16 + lm;
  const int koffA = hi ? 8 : 0;
  const float* hrow = h + (size_t)(m0 + lm) * HID;

  v8f c = {};
  for (int ks = 0; ks < 4; ++ks) {
    int kb = ks * 32;
    v16bf a;
#pragma unroll
    for (int j = 0; j < 16; ++j)
      a[j] = (__bf16)hrow[kb + koffA + ((j < 8) ? j : (8 + j))];
    v16bf b = load_b_pack(w0p, wv * 4 + ks, l);
    c = wmma_bf16(a, b, c);
  }
  const float bias = b0[n];
#pragma unroll
  for (int r = 0; r < 8; ++r) {
    int m = m0 + r + (hi ? 8 : 0);
    xout[(size_t)m * NF + n] = fmaxf(c[r] + bias, 0.f);
  }
}

// ---------------- fused edge kernel (64 edges / block) ----------------
__global__ __launch_bounds__(256) void edge_kernel(
    const float* __restrict__ x, const long long* __restrict__ eidx,
    const __bf16* __restrict__ eaBF, const __bf16* __restrict__ w1p,
    const float* __restrict__ b1, const __bf16* __restrict__ w2p,
    const float* __restrict__ b2, float* __restrict__ agg) {
  // sMsg overlays sEA: edge_attr tile is dead after the hidden GEMM barrier.
  __shared__ __align__(16) union {
    __bf16 ea[ET][HID];   // 16KB
    float msg[ET][NF];    // 16KB
  } sU;
  __shared__ __align__(16) __bf16 sH[ET][HID];  // 16KB
  __shared__ __align__(16) float sX[ET][NF];    // 16KB
  __shared__ int sDst[ET];

  const int e0 = blockIdx.x * ET;
  const int tid = threadIdx.x;
  const int wv = tid >> 5;
  const int l = tid & 31;
  const int lm = l & 15;
  const int hi = (l >= 16);
  const int koffA = hi ? 8 : 0;
  const int mo = hi ? 8 : 0;

  // Async gather: 4x GLOBAL_LOAD_ASYNC_TO_LDS_B128 per thread move scattered
  // 16B segments of x_src rows straight into LDS (tracked by ASYNCcnt).
#pragma unroll
  for (int q = 0; q < 4; ++q) {
    int idx = tid * 4 + q;   // 0..1023
    int row = idx >> 4;      // edge within tile
    int seg = idx & 15;      // 16B segment of the 256B row
    int src = (int)eidx[e0 + row];
    const float* gp = x + (size_t)src * NF + seg * 4;
    unsigned lds_off = (unsigned)(uintptr_t)&sX[row][seg * 4];
    asm volatile("global_load_async_to_lds_b128 %0, %1, off"
                 :
                 : "v"(lds_off), "v"((unsigned long long)(uintptr_t)gp)
                 : "memory");
  }

  // TDM: DMA the 16KB [64 x 128] bf16 edge_attr tile into LDS.
  if (wv == 0) {
    unsigned lds_base = (unsigned)(uintptr_t)&sU.ea[0][0];
    unsigned long long ga =
        (unsigned long long)(uintptr_t)eaBF + (unsigned long long)e0 * (HID * 2);
    const unsigned td0 = (ET * HID * 2) / 4;  // 4096 dwords
    uint32x4 g0;
    g0.x = 1u;                                // count = 1 (valid user D#)
    g0.y = lds_base;                          // lds_addr
    g0.z = (unsigned)(ga & 0xffffffffu);      // global_addr[31:0]
    g0.w = (unsigned)((ga >> 32) & 0x1ffffffu) | (2u << 30);  // addr hi | type=2
    uint32x8 g1;
    g1.s0 = (2u << 16);                // workgroup_mask=0, data_size=4B
    g1.s1 = (td0 & 0xffffu) << 16;     // tensor_dim0 lo16
    g1.s2 = (td0 >> 16) | (1u << 16);  // tensor_dim0 hi | tensor_dim1=1 lo
    g1.s3 = (td0 << 16);               // tensor_dim1 hi=0 | tile_dim0
    g1.s4 = 1u;                        // tile_dim1=1, tile_dim2=0
    g1.s5 = td0;                       // tensor_dim0_stride lo32
    g1.s6 = 0u;
    g1.s7 = 0u;
    asm volatile("tensor_load_to_lds %0, %1" : : "s"(g0), "s"(g1) : "memory");
    __builtin_amdgcn_s_wait_tensorcnt(0);
  }

  if (tid < ET) sDst[tid] = (int)eidx[N_EDGES + e0 + tid];
  __syncthreads();  // B1: sU.ea + sDst ready

  {  // hidden GEMM: wave wv -> columns [wv*16, wv*16+16), 4 M-tiles share B
    const int n = wv * 16 + lm;
    v8f c[4] = {};
    for (int ks = 0; ks < 4; ++ks) {
      v16bf b = load_b_pack(w1p, wv * 4 + ks, l);
#pragma unroll
      for (int mt = 0; mt < 4; ++mt) {
        v16bf a = load_a_lds(&sU.ea[mt * 16 + lm][0], ks * 32 + koffA);
        c[mt] = wmma_bf16(a, b, c[mt]);
      }
    }
    float bias = b1[n];
#pragma unroll
    for (int mt = 0; mt < 4; ++mt)
#pragma unroll
      for (int r = 0; r < 8; ++r)
        sH[mt * 16 + r + mo][n] = (__bf16)fmaxf(c[mt][r] + bias, 0.f);
  }
  asm volatile("s_wait_asynccnt 0x0" ::: "memory");  // sX resident in LDS
  __syncthreads();  // B2: sH + sX ready; sU.ea dead

  // A-fragments for all 4 M-tiles / 4 k-steps (reused for 32 column tiles)
  v16bf aT[4][4];
#pragma unroll
  for (int mt = 0; mt < 4; ++mt)
#pragma unroll
    for (int ks = 0; ks < 4; ++ks)
      aT[mt][ks] = load_a_lds(&sH[mt * 16 + lm][0], ks * 32 + koffA);

#pragma unroll
  for (int r = 0; r < 16; ++r) {  // zero sMsg (overlay region)
    int idx = r * 256 + tid;
    sU.msg[idx >> 6][idx & 63] = 0.f;
  }
  __syncthreads();  // B3: msg zeroed

  {  // W GEMM + fused contraction: each B fragment feeds 4 WMMAs
    const int ot = wv & 3;             // o-tile: cols [ot*16, ot*16+16)
    const int o0 = ot * 16;
    const int ibase = (wv >> 2) * 32;  // i half: [ibase, ibase+32)
    v8f macc[4] = {};
    for (int i = ibase; i < ibase + 32; ++i) {
      const int nt = i * 4 + ot;  // column tile (col0 = i*64 + o0)
      __builtin_prefetch(w2p + (((size_t)(nt + 4) * 4 * 32) << 4), 0, 0);
      v8f d[4] = {};
#pragma unroll
      for (int ks = 0; ks < 4; ++ks) {
        v16bf b = load_b_pack(w2p, nt * 4 + ks, l);
#pragma unroll
        for (int mt = 0; mt < 4; ++mt) d[mt] = wmma_bf16(aT[mt][ks], b, d[mt]);
      }
      const float bias = b2[i * NF + o0 + lm];
#pragma unroll
      for (int mt = 0; mt < 4; ++mt)
#pragma unroll
        for (int r = 0; r < 8; ++r)
          macc[mt][r] = fmaf(sX[mt * 16 + r + mo][i], d[mt][r] + bias, macc[mt][r]);
    }
#pragma unroll
    for (int mt = 0; mt < 4; ++mt)
#pragma unroll
      for (int r = 0; r < 8; ++r)
        atomicAdd(&sU.msg[mt * 16 + r + mo][o0 + lm], macc[mt][r]);
  }
  __syncthreads();  // B4: messages complete

#pragma unroll
  for (int r = 0; r < 16; ++r) {  // scatter (segment_sum)
    int idx = r * 256 + tid;
    int row = idx >> 6, col = idx & 63;
    atomicAdd(&agg[(size_t)sDst[row] * NF + col], sU.msg[row][col]);
  }
}

// ---------------- node update: x' = agg*inv_deg + x @ root_w + conv_b --------
__global__ __launch_bounds__(128) void node_update_kernel(
    const float* __restrict__ xin, const float* __restrict__ agg,
    const float* __restrict__ invdeg, const __bf16* __restrict__ rwp,
    const float* __restrict__ conv_b, float* __restrict__ xout) {
  const int m0 = blockIdx.x * 16;
  const int wv = threadIdx.x >> 5;
  const int l = threadIdx.x & 31;
  const int lm = l & 15;
  const int hi = (l >= 16);
  const int n = wv * 16 + lm;
  const int koffA = hi ? 8 : 0;
  const float* xrow = xin + (size_t)(m0 + lm) * NF;

  v8f c = {};
  for (int ks = 0; ks < 2; ++ks) {  // K = 64
    int kb = ks * 32;
    v16bf a;
#pragma unroll
    for (int j = 0; j < 16; ++j)
      a[j] = (__bf16)xrow[kb + koffA + ((j < 8) ? j : (8 + j))];
    v16bf b = load_b_pack(rwp, wv * 2 + ks, l);
    c = wmma_bf16(a, b, c);
  }
  const float bias = conv_b[n];
#pragma unroll
  for (int r = 0; r < 8; ++r) {
    int m = m0 + r + (hi ? 8 : 0);
    xout[(size_t)m * NF + n] = c[r] + agg[(size_t)m * NF + n] * invdeg[m] + bias;
  }
}

extern "C" void kernel_launch(void* const* d_in, const int* in_sizes, int n_in,
                              void* d_out, int out_size, void* d_ws, size_t ws_size,
                              hipStream_t stream) {
  (void)in_sizes; (void)n_in; (void)out_size; (void)ws_size;
  const float* h = (const float*)d_in[0];
  const long long* eidx = (const long long*)d_in[1];
  const float* edge_attr = (const float*)d_in[3];
  const float* lin0_w = (const float*)d_in[5];
  const float* lin0_b = (const float*)d_in[6];
  const float* nn_w1 = (const float*)d_in[7];
  const float* nn_b1 = (const float*)d_in[8];
  const float* nn_w2 = (const float*)d_in[9];
  const float* nn_b2 = (const float*)d_in[10];
  const float* root_w = (const float*)d_in[11];
  const float* conv_b = (const float*)d_in[12];
  float* out = (float*)d_out;

  char* ws = (char*)d_ws;
  size_t off = 0;
  auto alloc = [&](size_t bytes) -> void* {
    void* p = ws + off;
    off += (bytes + 255) & ~(size_t)255;
    return p;
  };
  float* xA = (float*)alloc((size_t)N_NODES * NF * 4);
  float* xB = (float*)alloc((size_t)N_NODES * NF * 4);
  float* agg = (float*)alloc((size_t)N_NODES * NF * 4);
  float* dinv = (float*)alloc((size_t)N_NODES * 4);
  __bf16* w0p = (__bf16*)alloc((size_t)HID * NF * 2);
  __bf16* w1p = (__bf16*)alloc((size_t)HID * HID * 2);
  __bf16* w2p = (__bf16*)alloc((size_t)HID * NF * NF * 2);
  __bf16* rwp = (__bf16*)alloc((size_t)NF * NF * 2);
  __bf16* eaBF = (__bf16*)alloc((size_t)N_EDGES * HID * 2);

  convert_weights_kernel<<<256, 256, 0, stream>>>(lin0_w, nn_w1, nn_w2, root_w,
                                                  w0p, w1p, w2p, rwp);
  convert_ea_kernel<<<8192, 256, 0, stream>>>(edge_attr, eaBF);
  zero_kernel<<<(N_NODES + 255) / 256, 256, 0, stream>>>(dinv, N_NODES);
  degree_kernel<<<(N_EDGES + 255) / 256, 256, 0, stream>>>(eidx, dinv);
  invdeg_kernel<<<(N_NODES + 255) / 256, 256, 0, stream>>>(dinv);
  lin0_kernel<<<N_NODES / 16, 128, 0, stream>>>(h, w0p, lin0_b, xA);

  const float* xc = xA;
  for (int it = 0; it < 2; ++it) {
    zero_kernel<<<(N_NODES * NF + 255) / 256, 256, 0, stream>>>(agg, N_NODES * NF);
    edge_kernel<<<N_EDGES / ET, 256, 0, stream>>>(xc, eidx, eaBF, w1p, nn_b1,
                                                  w2p, nn_b2, agg);
    float* xn = (it == 1) ? out : xB;
    node_update_kernel<<<N_NODES / 16, 128, 0, stream>>>(xc, agg, dinv, rwp,
                                                         conv_b, xn);
    xc = xn;
  }
}
